// Model_76879914598805
// MI455X (gfx1250) — matrix operand order repaired and checked by reading, not compiled, not run on hardware
//
#include <hip/hip_runtime.h>
#include <hip/hip_bf16.h>

typedef __attribute__((ext_vector_type(16))) _Float16 v16h;
typedef __attribute__((ext_vector_type(8)))  float    v8f;

#define H      128
#define MNBR   12
#define EE     20
#define NCONV  3
#define KTOT   288       // 128 self + 128 nbr + 32 (rbf padded 20->32)
#define NOUT   256       // 2*H gated output
#define AW     8         // atoms per workgroup
#define ROWS   (AW*MNBR) // 96 edge rows per workgroup
#define NKC    (KTOT/32) // 9 k-chunks
#define NCT    (NOUT/16) // 16 col-tiles
// packed B: per layer, [ct][kc][lane][16 halves]; layer stride in halves:
#define BPACK_LAYER (NCT*NKC*32*16)

__device__ __forceinline__ float sp_f(float x) {
    return (x > 20.f) ? x : log1pf(__expf(x));
}
__device__ __forceinline__ float sig_f(float x) {
    return 1.f / (1.f + __expf(-x));
}

union F16Frag { v16h v; uint4 q[2]; };
union H4Pack  { _Float16 h[4]; uint2 u; };

// ---------------------------------------------------------------------------
// Build per-layer combined weight, pre-swizzled into WMMA-B fragment layout:
//   logical B (KTOT x NOUT):
//     rows 0..127   = W_self (full_W rows 0..127)
//     rows 128..255 = W_nbr  (full_W rows 256..383)
//     rows 256..275 = filt_W @ W_edge (full_W rows 128..255), rows 276..287 = 0
//   bias2 = full_b + filt_b @ W_edge
//   packed[((ct*NKC+kc)*32 + lane)*16 + j]  where lane = (n&15) + 16*(kk>>4),
//   j = kk&15, ct = n>>4, kc = k>>5, kk = k&31.
// ---------------------------------------------------------------------------
__global__ void prep_weights_kernel(const float* __restrict__ filt_W,
                                    const float* __restrict__ filt_b,
                                    const float* __restrict__ full_W,
                                    const float* __restrict__ full_b,
                                    _Float16* __restrict__ Bp,
                                    float* __restrict__ bias2) {
    const int l = blockIdx.x;
    const int n = threadIdx.x; // 0..255 (output column)
    const float* W = full_W + (size_t)l * (3 * H) * NOUT;
    _Float16* B = Bp + (size_t)l * BPACK_LAYER;

    const int ct = n >> 4;
    const int lo = n & 15;

    for (int k = 0; k < KTOT; ++k) {
        float v;
        if (k < H)            v = W[k * NOUT + n];                 // self
        else if (k < 2 * H)   v = W[(2 * H + (k - H)) * NOUT + n]; // nbr
        else if (k < 2 * H + EE) {                                 // combined edge
            const int e = k - 2 * H;
            const float* FW = filt_W + (size_t)l * EE * H + e * H;
            float s = 0.f;
            for (int j = 0; j < H; ++j)
                s += FW[j] * W[(H + j) * NOUT + n];
            v = s;
        } else v = 0.f;                                            // pad
        const int kc = k >> 5;
        const int kk = k & 31;
        const int lane = lo + ((kk >> 4) << 4);
        const int j = kk & 15;
        B[((ct * NKC + kc) * 32 + lane) * 16 + j] = (_Float16)v;
    }

    const float* fb = filt_b + (size_t)l * H;
    float s = 0.f;
    for (int j = 0; j < H; ++j)
        s += fb[j] * W[(H + j) * NOUT + n];
    bias2[l * NOUT + n] = full_b[l * NOUT + n] + s;
}

// ---------------------------------------------------------------------------
// nodes = atoms_embed @ embed_W + embed_b
// ---------------------------------------------------------------------------
__global__ void embed_kernel(const float* __restrict__ atoms_embed,
                             const float* __restrict__ embed_W,
                             const float* __restrict__ embed_b,
                             float* __restrict__ nodes) {
    const int a = blockIdx.x;
    const int h = threadIdx.x;
    float s = embed_b[h];
#pragma unroll
    for (int j = 0; j < 13; ++j)
        s += atoms_embed[a * 13 + j] * embed_W[j * H + h];
    nodes[(size_t)a * H + h] = s;
}

// ---------------------------------------------------------------------------
// Fused conv layer. 8 atoms (96 edge rows) per workgroup.
// A (96x288 f16) in LDS; B fragments loaded straight from pre-swizzled global
// (L2-resident, lane-coalesced b128). Barrier-free WMMA main loop; all 6 A
// fragments of a k-chunk preloaded into distinct registers so the scheduler
// can overlap LDS latency with the matrix pipe (partial dscnt waits).
// ---------------------------------------------------------------------------
__global__ __launch_bounds__(256) void conv_kernel(
        const float* __restrict__ nodes_in,
        float* __restrict__ nodes_out,
        const int* __restrict__ nbrs_idx,
        const float* __restrict__ nbrs_fea,
        const _Float16* __restrict__ Bp,     // layer slice, packed fragments
        const float* __restrict__ bias2) {   // layer slice, NOUT

    __shared__ __align__(16) _Float16 Ah[ROWS * KTOT]; // 55296 B
    __shared__ int nbrC[ROWS];

    const int tid = threadIdx.x;
    const int a0  = blockIdx.x * AW;

    if (tid < ROWS)
        nbrC[tid] = nbrs_idx[(a0 + tid / MNBR) * MNBR + tid % MNBR];
    __syncthreads();

    // Self cols 0..127: one float4 chunk per thread, replicated to 12 rows
    {
        const int la = tid >> 5;          // 0..7
        const int k4 = (tid & 31) << 2;   // 0..124
        const float4 v = *(const float4*)&nodes_in[(size_t)(a0 + la) * H + k4];
        H4Pack p;
        p.h[0] = (_Float16)v.x; p.h[1] = (_Float16)v.y;
        p.h[2] = (_Float16)v.z; p.h[3] = (_Float16)v.w;
#pragma unroll
        for (int m = 0; m < MNBR; ++m)
            *(uint2*)&Ah[(la * MNBR + m) * KTOT + k4] = p.u;
    }
    // Gathered neighbor cols 128..255: float4 loads
    for (int flat = tid; flat < ROWS * 32; flat += 256) {
        const int row = flat >> 5;
        const int k4  = (flat & 31) << 2;
        const float4 v = *(const float4*)&nodes_in[(size_t)nbrC[row] * H + k4];
        H4Pack p;
        p.h[0] = (_Float16)v.x; p.h[1] = (_Float16)v.y;
        p.h[2] = (_Float16)v.z; p.h[3] = (_Float16)v.w;
        *(uint2*)&Ah[row * KTOT + H + k4] = p.u;
    }
    // RBF cols 256..287 (20 real + 12 zero pad)
    if (tid < ROWS) {
        const int la = tid / MNBR, m = tid % MNBR;
        const float d   = nbrs_fea[(a0 + la) * MNBR + m];
        const float env = 0.5f * (__cosf(3.14159265358979f * d * 0.125f) + 1.f);
        const float inv = 1.f / d;
        const bool  in  = d < 8.f;
        for (int e = 0; e < EE; ++e) {
            const float f = (float)(e + 1) * (3.14159265358979f * 0.125f);
            const float v = in ? __sinf(d * f) * inv * env : 0.f;
            Ah[tid * KTOT + 2 * H + e] = (_Float16)v;
        }
        for (int e = EE; e < 32; ++e)
            Ah[tid * KTOT + 2 * H + e] = (_Float16)0.f;
    }
    __syncthreads();

    const int lane = tid & 31;
    const int w    = tid >> 5;            // wave id 0..7
    const int lrow = lane & 15;
    const int hi   = (lane >> 4) & 1;
    const int n0   = w * 16;              // filter col-tile; core = n0+128
    const int ctF  = w;
    const int ctC  = w + 8;

    v8f accF[6], accC[6];
#pragma unroll
    for (int rt = 0; rt < 6; ++rt) { accF[rt] = (v8f){}; accC[rt] = (v8f){}; }

#pragma unroll
    for (int kc = 0; kc < NKC; ++kc) {
        const int kb = kc * 32;
        // B fragments straight from global (pre-swizzled, lane-coalesced)
        F16Frag bF, bC;
        {
            const _Float16* pF = Bp + ((size_t)(ctF * NKC + kc) * 32 + lane) * 16;
            const _Float16* pC = Bp + ((size_t)(ctC * NKC + kc) * 32 + lane) * 16;
            bF.q[0] = *(const uint4*)pF;
            bF.q[1] = *(const uint4*)(pF + 8);
            bC.q[0] = *(const uint4*)pC;
            bC.q[1] = *(const uint4*)(pC + 8);
        }
        // Preload ALL six A fragments into distinct registers first
        F16Frag af[6];
#pragma unroll
        for (int rt = 0; rt < 6; ++rt) {
            const _Float16* pA = &Ah[(rt * 16 + lrow) * KTOT + kb + hi * 16];
            af[rt].q[0] = *(const uint4*)pA;
            af[rt].q[1] = *(const uint4*)(pA + 8);
        }
        // Then the 12 back-to-back WMMAs
#pragma unroll
        for (int rt = 0; rt < 6; ++rt) {
            accF[rt] = __builtin_amdgcn_wmma_f32_16x16x32_f16(
                false, af[rt].v, false, bF.v, (short)0, accF[rt], false, false);
            accC[rt] = __builtin_amdgcn_wmma_f32_16x16x32_f16(
                false, af[rt].v, false, bC.v, (short)0, accC[rt], false, false);
        }
    }
    __syncthreads();

    // Epilogue: sigmoid(filter) * softplus(core), stage to LDS (overlay Ah)
    float* prod = (float*)Ah; // 96*128*4 = 49152 B <= sizeof(Ah) = 55296 B
    const float bf = bias2[n0 + lrow];
    const float bc = bias2[128 + n0 + lrow];
#pragma unroll
    for (int rt = 0; rt < 6; ++rt) {
#pragma unroll
        for (int r = 0; r < 8; ++r) {
            const float fv = accF[rt][r] + bf;
            const float cv = accC[rt][r] + bc;
            const int row  = rt * 16 + r + hi * 8;
            prod[row * H + (n0 + lrow)] = sig_f(fv) * sp_f(cv);
        }
    }
    __syncthreads();

    // Reduce over 12 neighbors, nodes_out = softplus(nodes_in + agg)
    for (int flat = tid; flat < AW * H; flat += 256) {
        const int la = flat >> 7;
        const int h  = flat & (H - 1);
        float s = 0.f;
#pragma unroll
        for (int m = 0; m < MNBR; ++m)
            s += prod[(la * MNBR + m) * H + h];
        const float x = nodes_in[(size_t)(a0 + la) * H + h] + s;
        nodes_out[(size_t)(a0 + la) * H + h] = sp_f(x);
    }
}

// ---------------------------------------------------------------------------
// Per-crystal mean pool + softplus(crys@c2f_W + b) @ out_W + out_b
// ---------------------------------------------------------------------------
__global__ void head_kernel(const float* __restrict__ nodes,
                            const int* __restrict__ num_atoms,
                            const float* __restrict__ c2f_W,
                            const float* __restrict__ c2f_b,
                            const float* __restrict__ out_W,
                            const float* __restrict__ out_b,
                            float* __restrict__ out) {
    __shared__ float sm[H];
    __shared__ float sv[H];
    __shared__ int s_off, s_cnt;
    const int b = blockIdx.x;
    const int tid = threadIdx.x; // 128
    if (tid == 0) {
        int o = 0;
        for (int j = 0; j < b; ++j) o += num_atoms[j];
        s_off = o;
        s_cnt = num_atoms[b];
    }
    __syncthreads();
    const int off = s_off, cnt = s_cnt;
    float s = 0.f;
    for (int i = 0; i < cnt; ++i)
        s += nodes[(size_t)(off + i) * H + tid];
    sm[tid] = s / (float)cnt;
    __syncthreads();
    float acc = c2f_b[tid];
    for (int j = 0; j < H; ++j)
        acc += sm[j] * c2f_W[j * H + tid];
    sv[tid] = sp_f(acc) * out_W[tid];
    __syncthreads();
    for (int st = 64; st > 0; st >>= 1) {
        if (tid < st) sv[tid] += sv[tid + st];
        __syncthreads();
    }
    if (tid == 0) out[b] = sv[0] + out_b[0];
}

// ---------------------------------------------------------------------------
extern "C" void kernel_launch(void* const* d_in, const int* in_sizes, int n_in,
                              void* d_out, int out_size, void* d_ws, size_t ws_size,
                              hipStream_t stream) {
    const float* atoms_embed = (const float*)d_in[0];
    const float* nbrs_fea    = (const float*)d_in[1];
    const float* embed_W     = (const float*)d_in[2];
    const float* embed_b     = (const float*)d_in[3];
    const float* filt_W      = (const float*)d_in[4];
    const float* filt_b      = (const float*)d_in[5];
    const float* full_W      = (const float*)d_in[6];
    const float* full_b      = (const float*)d_in[7];
    const float* c2f_W       = (const float*)d_in[8];
    const float* c2f_b       = (const float*)d_in[9];
    const float* out_W       = (const float*)d_in[10];
    const float* out_b       = (const float*)d_in[11];
    const int*   nbrs_idx    = (const int*)d_in[12];
    const int*   num_atoms   = (const int*)d_in[13];

    const int N = in_sizes[0] / 13;
    const int B = in_sizes[13];

    // workspace layout
    const size_t nodesElems = (size_t)N * H;
    float*     nodes0 = (float*)d_ws;
    float*     nodes1 = nodes0 + nodesElems;
    _Float16*  Bp     = (_Float16*)(nodes1 + nodesElems);
    float*     bias2  = (float*)((char*)Bp + (size_t)NCONV * BPACK_LAYER * sizeof(_Float16));

    prep_weights_kernel<<<NCONV, 256, 0, stream>>>(filt_W, filt_b, full_W, full_b, Bp, bias2);
    embed_kernel<<<N, H, 0, stream>>>(atoms_embed, embed_W, embed_b, nodes0);

    const int nblk = N / AW;
    conv_kernel<<<nblk, 256, 0, stream>>>(nodes0, nodes1, nbrs_idx, nbrs_fea,
                                          Bp + (size_t)0 * BPACK_LAYER, bias2 + 0 * NOUT);
    conv_kernel<<<nblk, 256, 0, stream>>>(nodes1, nodes0, nbrs_idx, nbrs_fea,
                                          Bp + (size_t)1 * BPACK_LAYER, bias2 + 1 * NOUT);
    conv_kernel<<<nblk, 256, 0, stream>>>(nodes0, nodes1, nbrs_idx, nbrs_fea,
                                          Bp + (size_t)2 * BPACK_LAYER, bias2 + 2 * NOUT);

    head_kernel<<<B, H, 0, stream>>>(nodes1, num_atoms, c2f_W, c2f_b, out_W, out_b,
                                     (float*)d_out);
}
